// AttentionLayerEnhance_64518998720726
// MI455X (gfx1250) — matrix operand order repaired and checked by reading, not compiled, not run on hardware
//
#include <hip/hip_runtime.h>
#include <cstdint>
#include <cstddef>

#define BSZ 4
#define LEN 2048
#define DM  512
#define NH  8
#define DK  64
#define WIN 16   // half-window: band is s in [l-16, l]

typedef unsigned short ushort_t;
typedef __attribute__((ext_vector_type(16))) __bf16          v16bf;
typedef __attribute__((ext_vector_type(8)))  float           v8f;
typedef __attribute__((ext_vector_type(4)))  unsigned int    u32x4;
typedef __attribute__((ext_vector_type(4)))  int             i32x4;
typedef __attribute__((ext_vector_type(4)))  float           f32x4;
typedef __attribute__((ext_vector_type(4)))  unsigned short  u16x4;

union Frag { u32x4 u[2]; v16bf v; };

// ---- async global->LDS copy support (gfx1250 GLOBAL_LOAD_ASYNC_TO_LDS) ----
#if defined(__has_builtin)
# if __has_builtin(__builtin_amdgcn_global_load_async_to_lds_b128)
#  define HAVE_ASYNC_LDS 1
# endif
#endif
#ifndef HAVE_ASYNC_LDS
# define HAVE_ASYNC_LDS 0
#endif

#if HAVE_ASYNC_LDS
typedef __attribute__((address_space(1))) i32x4 as1_i32x4;
typedef __attribute__((address_space(3))) i32x4 as3_i32x4;
__device__ __forceinline__ void async_copy_b128(const ushort_t* g, ushort_t* l) {
  __builtin_amdgcn_global_load_async_to_lds_b128(
      (as1_i32x4*)const_cast<ushort_t*>(g), (as3_i32x4*)l, 0, 0);
}
# if __has_builtin(__builtin_amdgcn_s_wait_asynccnt)
#  define WAIT_ASYNC0() __builtin_amdgcn_s_wait_asynccnt(0)
# else
#  define WAIT_ASYNC0() asm volatile("s_wait_asynccnt 0x0" ::: "memory")
# endif
#endif

__device__ __forceinline__ ushort_t f2bf(float f) {
  union { float f; unsigned u; } v; v.f = f;
  unsigned r = v.u + 0x7FFFu + ((v.u >> 16) & 1u);   // round-to-nearest-even
  return (ushort_t)(r >> 16);
}
__device__ __forceinline__ float bf2f(ushort_t h) {
  union { unsigned u; float f; } v; v.u = ((unsigned)h) << 16;
  return v.f;
}

// ---------------------------------------------------------------------------
// fp32 -> bf16 elementwise cast (vectorized x4)
// ---------------------------------------------------------------------------
__global__ __launch_bounds__(256) void cast_bf16x4_kernel(
    const float* __restrict__ x, ushort_t* __restrict__ y, int n4)
{
  int i = blockIdx.x * blockDim.x + threadIdx.x;
  if (i < n4) {
    f32x4 f = *(const f32x4*)(x + (size_t)i * 4);
    u16x4 o;
    o[0] = f2bf(f[0]); o[1] = f2bf(f[1]); o[2] = f2bf(f[2]); o[3] = f2bf(f[3]);
    *(u16x4*)(y + (size_t)i * 4) = o;
  }
}

// ---------------------------------------------------------------------------
// W[512,512] fp32 -> Wt[n][k] = W[k][n] bf16 (B-operand layout for WMMA)
// ---------------------------------------------------------------------------
__global__ __launch_bounds__(256) void transpose_cast_kernel(
    const float* __restrict__ W, ushort_t* __restrict__ Wt)
{
  __shared__ float tile[32][33];
  const int bx = blockIdx.x * 32;  // n
  const int by = blockIdx.y * 32;  // k
  const int tx = threadIdx.x, ty = threadIdx.y;
  for (int i = ty; i < 32; i += 8)
    tile[i][tx] = W[(size_t)(by + i) * DM + bx + tx];
  __syncthreads();
  for (int i = ty; i < 32; i += 8)
    Wt[(size_t)(bx + i) * DM + by + tx] = f2bf(tile[tx][i]);
}

// ---------------------------------------------------------------------------
// Tiled bf16 WMMA GEMM: Y[8192,512] = X[8192,512] @ W[512,512] + bias
// Wt holds W transposed (row n = column n of W, K-contiguous).
// Block tile 128x128, K-step 32, 8 waves (wave32) in 4(M)x2(N) grid,
// each wave computes 32x64 via 2x4 fragments of v_wmma_f32_16x16x32_bf16.
// Double-buffered LDS filled by GLOBAL_LOAD_ASYNC_TO_LDS_B128 (ASYNCcnt),
// one barrier per K-step; falls back to register staging if the async
// builtin is unavailable.
// ---------------------------------------------------------------------------
__global__ __launch_bounds__(256) void gemm_bf16_wmma(
    const ushort_t* __restrict__ X,
    const ushort_t* __restrict__ Wt,
    const float*    __restrict__ bias,
    ushort_t* __restrict__ outB,   // bf16 output (or null)
    float*    __restrict__ outF)   // fp32 output (or null)
{
  __shared__ ushort_t ldsA[2][128 * 40];   // 32-elem rows padded to 40 (80 B)
  __shared__ ushort_t ldsB[2][128 * 40];

  const int tid   = threadIdx.x;
  const int lane  = tid & 31;
  const int wid   = tid >> 5;
  const int wm    = wid >> 1;          // 0..3
  const int wn    = wid & 1;           // 0..1
  const int tileM = blockIdx.y * 128;
  const int tileN = blockIdx.x * 128;

  // staging: 512 chunks of 16 B per tile, 2 per thread
  const int r0 = tid >> 2;
  const int r1 = r0 + 64;
  const int c0 = (tid & 3) << 3;       // element offset within 32-wide slice

  const ushort_t* gA0 = X  + (size_t)(tileM + r0) * DM + c0;
  const ushort_t* gA1 = X  + (size_t)(tileM + r1) * DM + c0;
  const ushort_t* gB0 = Wt + (size_t)(tileN + r0) * DM + c0;
  const ushort_t* gB1 = Wt + (size_t)(tileN + r1) * DM + c0;
  const int lA0 = r0 * 40 + c0;
  const int lA1 = r1 * 40 + c0;

  v8f acc[2][4];
  v8f zf = {};
#pragma unroll
  for (int i = 0; i < 2; ++i)
#pragma unroll
    for (int j = 0; j < 4; ++j) acc[i][j] = zf;

  // prologue: stage K-slice 0 into buffer 0
#if HAVE_ASYNC_LDS
  async_copy_b128(gA0, &ldsA[0][lA0]);
  async_copy_b128(gA1, &ldsA[0][lA1]);
  async_copy_b128(gB0, &ldsB[0][lA0]);
  async_copy_b128(gB1, &ldsB[0][lA1]);
#else
  *(u32x4*)&ldsA[0][lA0] = *(const u32x4*)gA0;
  *(u32x4*)&ldsA[0][lA1] = *(const u32x4*)gA1;
  *(u32x4*)&ldsB[0][lA0] = *(const u32x4*)gB0;
  *(u32x4*)&ldsB[0][lA1] = *(const u32x4*)gB1;
#endif

  const int laneR = lane & 15;
  const int laneH = lane >> 4;
  // A frag: lane half 0 -> K 0..7 & 16..23 ; half 1 -> K 8..15 & 24..31
  const int aOff0 = (wm * 32 + laneR) * 40 + laneH * 8;
  // B frag: lane half 0 -> K 0..7 & 16..23 ; half 1 -> K 8..15 & 24..31
  const int bOff0 = (wn * 64 + laneR) * 40 + laneH * 8;

#pragma unroll 1
  for (int kt = 0; kt < 16; ++kt) {
    const int cur = kt & 1;
    const int nxt = cur ^ 1;

#if HAVE_ASYNC_LDS
    WAIT_ASYNC0();          // my slice-kt copies have landed in LDS[cur]
    __syncthreads();        // everyone's landed; buffer[nxt] free everywhere
    if (kt < 15) {          // stream slice kt+1 while we do the WMMAs
      const int ko = (kt + 1) * 32;
      async_copy_b128(gA0 + ko, &ldsA[nxt][lA0]);
      async_copy_b128(gA1 + ko, &ldsA[nxt][lA1]);
      async_copy_b128(gB0 + ko, &ldsB[nxt][lA0]);
      async_copy_b128(gB1 + ko, &ldsB[nxt][lA1]);
    }
#else
    __syncthreads();
#endif
    if (kt < 14) {
      const int kp = (kt + 2) * 32;
      __builtin_prefetch((const void*)(gA0 + kp), 0, 1);
      __builtin_prefetch((const void*)(gB0 + kp), 0, 1);
    }

    Frag fa[2], fb[4];
#pragma unroll
    for (int fm = 0; fm < 2; ++fm) {
      const int o = aOff0 + fm * 16 * 40;
      fa[fm].u[0] = *(const u32x4*)&ldsA[cur][o];
      fa[fm].u[1] = *(const u32x4*)&ldsA[cur][o + 16];
    }
#pragma unroll
    for (int fn = 0; fn < 4; ++fn) {
      const int o = bOff0 + fn * 16 * 40;
      fb[fn].u[0] = *(const u32x4*)&ldsB[cur][o];
      fb[fn].u[1] = *(const u32x4*)&ldsB[cur][o + 16];
    }

#if !HAVE_ASYNC_LDS
    if (kt < 15) {          // register-staged fallback: fill buffer[nxt]
      const int ko = (kt + 1) * 32;
      u32x4 na0 = *(const u32x4*)(gA0 + ko);
      u32x4 na1 = *(const u32x4*)(gA1 + ko);
      u32x4 nb0 = *(const u32x4*)(gB0 + ko);
      u32x4 nb1 = *(const u32x4*)(gB1 + ko);
      *(u32x4*)&ldsA[nxt][lA0] = na0;
      *(u32x4*)&ldsA[nxt][lA1] = na1;
      *(u32x4*)&ldsB[nxt][lA0] = nb0;
      *(u32x4*)&ldsB[nxt][lA1] = nb1;
    }
#endif

#pragma unroll
    for (int fm = 0; fm < 2; ++fm)
#pragma unroll
      for (int fn = 0; fn < 4; ++fn)
        acc[fm][fn] = __builtin_amdgcn_wmma_f32_16x16x32_bf16(
            false, fa[fm].v, false, fb[fn].v, (short)0, acc[fm][fn],
            false, false);
  }

  // epilogue: C/D layout -> VGPR r: M = r + 8*laneH, N = laneR
  if (outB) {
#pragma unroll
    for (int fm = 0; fm < 2; ++fm)
#pragma unroll
      for (int fn = 0; fn < 4; ++fn) {
        const int col = tileN + wn * 64 + fn * 16 + laneR;
        const float bv = bias[col];
#pragma unroll
        for (int r = 0; r < 8; ++r) {
          const int row = tileM + wm * 32 + fm * 16 + laneH * 8 + r;
          outB[(size_t)row * DM + col] = f2bf(acc[fm][fn][r] + bv);
        }
      }
  } else {
#pragma unroll
    for (int fm = 0; fm < 2; ++fm)
#pragma unroll
      for (int fn = 0; fn < 4; ++fn) {
        const int col = tileN + wn * 64 + fn * 16 + laneR;
        const float bv = bias[col];
#pragma unroll
        for (int r = 0; r < 8; ++r) {
          const int row = tileM + wm * 32 + fm * 16 + laneH * 8 + r;
          outF[(size_t)row * DM + col] = acc[fm][fn][r] + bv;
        }
      }
  }
}

// ---------------------------------------------------------------------------
// Banded attention: window(±16) ∩ causal leaves <=17 keys per query.
// One thread per (b,h,l): scores * exp(-g*rel), softmax over band,
// scatter probs into the (pre-zeroed) dense attn row, accumulate context.
// ---------------------------------------------------------------------------
__global__ __launch_bounds__(256) void attn_band_kernel(
    const ushort_t* __restrict__ Q, const ushort_t* __restrict__ K,
    const ushort_t* __restrict__ V, const float* __restrict__ gamma,
    float* __restrict__ attn, ushort_t* __restrict__ ctx)
{
  const int idx = blockIdx.x * blockDim.x + threadIdx.x;
  if (idx >= BSZ * NH * LEN) return;
  const int l = idx & (LEN - 1);
  const int h = (idx >> 11) & (NH - 1);
  const int b = idx >> 14;
  const float g = gamma[0];
  const float scale = 0.04419417382415922f;  // 1/sqrt(512)

  const size_t qoff = ((size_t)(b * LEN + l)) * DM + h * DK;
  float q[DK];
#pragma unroll
  for (int c = 0; c < 8; ++c) {
    u32x4 u = *(const u32x4*)(Q + qoff + c * 8);
    const ushort_t* p = (const ushort_t*)&u;
#pragma unroll
    for (int j = 0; j < 8; ++j) q[c * 8 + j] = bf2f(p[j]);
  }

  const int s0  = (l - WIN) > 0 ? (l - WIN) : 0;
  const int cnt = l - s0 + 1;                 // <= 17
  float sc[2 * WIN + 1];
  float mx = -3.0e38f;
  for (int i = 0; i < cnt; ++i) {
    const int s = s0 + i;
    const size_t koff = ((size_t)(b * LEN + s)) * DM + h * DK;
    float d = 0.f;
#pragma unroll
    for (int c = 0; c < 8; ++c) {
      u32x4 u = *(const u32x4*)(K + koff + c * 8);
      const ushort_t* p = (const ushort_t*)&u;
#pragma unroll
      for (int j = 0; j < 8; ++j) d += q[c * 8 + j] * bf2f(p[j]);
    }
    const float rel = (float)(l - s);
    const float v = d * scale * __expf(-g * rel);
    sc[i] = v;
    mx = fmaxf(mx, v);
  }
  float sum = 0.f;
  for (int i = 0; i < cnt; ++i) { const float e = __expf(sc[i] - mx); sc[i] = e; sum += e; }
  const float inv = 1.0f / sum;

  float cv[DK];
#pragma unroll
  for (int j = 0; j < DK; ++j) cv[j] = 0.f;

  float* arow = attn + ((size_t)((b * NH + h) * LEN + l)) * LEN;
  for (int i = 0; i < cnt; ++i) {
    const float a = sc[i] * inv;
    arow[s0 + i] = a;
    const size_t voff = ((size_t)(b * LEN + s0 + i)) * DM + h * DK;
#pragma unroll
    for (int c = 0; c < 8; ++c) {
      u32x4 u = *(const u32x4*)(V + voff + c * 8);
      const ushort_t* p = (const ushort_t*)&u;
#pragma unroll
      for (int j = 0; j < 8; ++j) cv[c * 8 + j] += a * bf2f(p[j]);
    }
  }
#pragma unroll
  for (int j = 0; j < DK; ++j) ctx[qoff + j] = f2bf(cv[j]);
}

// ---------------------------------------------------------------------------
// Host launcher
// ---------------------------------------------------------------------------
extern "C" void kernel_launch(void* const* d_in, const int* in_sizes, int n_in,
                              void* d_out, int out_size, void* d_ws, size_t ws_size,
                              hipStream_t stream)
{
  (void)in_sizes; (void)n_in; (void)out_size; (void)ws_size;

  const float* queries = (const float*)d_in[0];
  const float* keys    = (const float*)d_in[1];
  const float* values  = (const float*)d_in[2];
  // d_in[3] = causal mask (analytic, unused)
  const float* Wq = (const float*)d_in[4];
  const float* bq = (const float*)d_in[5];
  const float* Wk = (const float*)d_in[6];
  const float* bk = (const float*)d_in[7];
  const float* Wv = (const float*)d_in[8];
  const float* bv = (const float*)d_in[9];
  const float* Wo = (const float*)d_in[10];
  const float* bo = (const float*)d_in[11];
  const float* gamma = (const float*)d_in[12];

  const size_t NE = (size_t)BSZ * LEN * DM;   // 4,194,304 activation elems
  const size_t WE = (size_t)DM * DM;          //   262,144 weight elems

  ushort_t* w   = (ushort_t*)d_ws;
  ushort_t* Xq  = w;
  ushort_t* Xk  = Xq + NE;
  ushort_t* Xv  = Xk + NE;
  ushort_t* Qb  = Xv + NE;
  ushort_t* Kb  = Qb + NE;
  ushort_t* Vb  = Kb + NE;
  ushort_t* Cx  = Vb + NE;
  ushort_t* WqT = Cx + NE;
  ushort_t* WkT = WqT + WE;
  ushort_t* WvT = WkT + WE;
  ushort_t* WoT = WvT + WE;

  float* out  = (float*)d_out;
  float* attn = out + NE;   // [B,H,L,S] fp32

  // zero the dense attention output (band writes fill the rest)
  (void)hipMemsetAsync(attn, 0,
      (size_t)BSZ * NH * LEN * (size_t)LEN * sizeof(float), stream);

  {
    const int n4 = (int)(NE / 4);
    dim3 g((n4 + 255) / 256), blk(256);
    cast_bf16x4_kernel<<<g, blk, 0, stream>>>(queries, Xq, n4);
    cast_bf16x4_kernel<<<g, blk, 0, stream>>>(keys,    Xk, n4);
    cast_bf16x4_kernel<<<g, blk, 0, stream>>>(values,  Xv, n4);
  }
  {
    dim3 g(16, 16), blk(32, 8);
    transpose_cast_kernel<<<g, blk, 0, stream>>>(Wq, WqT);
    transpose_cast_kernel<<<g, blk, 0, stream>>>(Wk, WkT);
    transpose_cast_kernel<<<g, blk, 0, stream>>>(Wv, WvT);
    transpose_cast_kernel<<<g, blk, 0, stream>>>(Wo, WoT);
  }
  {
    dim3 g(4, 64), blk(256);
    gemm_bf16_wmma<<<g, blk, 0, stream>>>(Xq, WqT, bq, Qb, nullptr);
    gemm_bf16_wmma<<<g, blk, 0, stream>>>(Xk, WkT, bk, Kb, nullptr);
    gemm_bf16_wmma<<<g, blk, 0, stream>>>(Xv, WvT, bv, Vb, nullptr);
  }
  {
    dim3 g((BSZ * NH * LEN + 255) / 256), blk(256);
    attn_band_kernel<<<g, blk, 0, stream>>>(Qb, Kb, Vb, gamma, attn, Cx);
  }
  {
    dim3 g(4, 64), blk(256);
    gemm_bf16_wmma<<<g, blk, 0, stream>>>(Cx, WoT, bo, nullptr, out);
  }
}
